// DGCNNDecoder_46127948759241
// MI455X (gfx1250) — matrix operand order repaired and checked by reading, not compiled, not run on hardware
//
#include <hip/hip_runtime.h>

#define BS 4
#define NX 8192
#define NY 1024
#define KNN 20
#define CD 24
#define HD 128
#define NB 5

typedef __attribute__((ext_vector_type(16))) __bf16 v16bf;
typedef __attribute__((ext_vector_type(8)))  float  v8f;
typedef __attribute__((ext_vector_type(4)))  unsigned int u32x4;

#define WMMA_BF16(A, B, C) \
  __builtin_amdgcn_wmma_f32_16x16x32_bf16(false, (A), false, (B), (short)0, (C), false, false)

// ---------- CDNA5 async global->LDS helpers (ASYNCcnt-tracked DMA) ----------
__device__ __forceinline__ unsigned lds_off(const void* p) {
  return (unsigned)(size_t)p;  // generic LDS pointer: addr[31:0] is the LDS byte offset
}
__device__ __forceinline__ void async_b128(unsigned ldsOff, const void* g) {
  asm volatile("global_load_async_to_lds_b128 %0, %1, off" :: "v"(ldsOff), "v"(g) : "memory");
}
__device__ __forceinline__ void wait_async0() {
  asm volatile("s_wait_asynccnt 0x0" ::: "memory");
}

// ---------- fragment loaders (contiguous k order: element s of half g is k = 16*g + s, in A and B) ----------
__device__ __forceinline__ v16bf load_a_frag(const __bf16* rowbase, int m, int g, int stride) {
  const __bf16* rp = rowbase + m * stride;
  v16bf a;
#pragma unroll
  for (int j = 0; j < 8; ++j) a[j] = rp[g * 16 + j];
#pragma unroll
  for (int j = 0; j < 8; ++j) a[j + 8] = rp[g * 16 + 8 + j];
  return a;
}
__device__ __forceinline__ v16bf load_b_frag(const __bf16* wT, int n, int g, int k0, int stride) {
  const __bf16* rp = wT + (size_t)n * stride + k0 + g * 16;
  v16bf b;
#pragma unroll
  for (int j = 0; j < 16; ++j) b[j] = rp[j];
  return b;
}
__device__ __forceinline__ v8f zero_v8f() {
  v8f z = {0.f, 0.f, 0.f, 0.f, 0.f, 0.f, 0.f, 0.f};
  return z;
}

// ---------- weight convert: src [K][N] f32 -> dst [Np][Kp] bf16 (transposed + zero pad) ----------
__global__ void cvt_t_kernel(const float* __restrict__ src, __bf16* __restrict__ dst,
                             int K, int N, int Kp, int Np) {
  int t = blockIdx.x * 256 + threadIdx.x;
  if (t >= Np * Kp) return;
  int n = t / Kp, k = t % Kp;
  float v = (n < N && k < K) ? src[(size_t)k * N + n] : 0.f;
  dst[t] = (__bf16)v;
}

// ---------- KNN: one thread per query, pc tile in LDS, register insertion sort ----------
__global__ __launch_bounds__(256) void knn_kernel(const float* __restrict__ p,
                                                  const float* __restrict__ pc,
                                                  int* __restrict__ idx) {
  __shared__ float spc[NY * 3];
  const int blocksPerBatch = NX / 256;
  int b = blockIdx.x / blocksPerBatch;
  int n0 = (blockIdx.x % blocksPerBatch) * 256;
  for (int e = threadIdx.x; e < NY * 3; e += 256) spc[e] = pc[(size_t)b * NY * 3 + e];
  __syncthreads();
  int n = n0 + threadIdx.x;
  size_t pbase = ((size_t)b * NX + n) * 3;
  float px = p[pbase + 0], py = p[pbase + 1], pz = p[pbase + 2];
  float bd[KNN];
  int bi_[KNN];
#pragma unroll
  for (int j = 0; j < KNN; ++j) { bd[j] = 3.4e38f; bi_[j] = 0; }
  for (int mm = 0; mm < NY; ++mm) {
    float dx = spc[mm * 3 + 0] - px;
    float dy = spc[mm * 3 + 1] - py;
    float dz = spc[mm * 3 + 2] - pz;
    float d = dx * dx + dy * dy + dz * dz;
    if (d < bd[KNN - 1]) {
      bd[KNN - 1] = d; bi_[KNN - 1] = mm;
#pragma unroll
      for (int j = KNN - 1; j > 0; --j) {
        if (bd[j] < bd[j - 1]) {
          float tf = bd[j]; bd[j] = bd[j - 1]; bd[j - 1] = tf;
          int ti = bi_[j]; bi_[j] = bi_[j - 1]; bi_[j - 1] = ti;
        }
      }
    }
  }
#pragma unroll
  for (int j = 0; j < KNN; ++j) idx[(((size_t)b * NX + n) * KNN) + j] = bi_[j];
}

// ---------- gather: build X [BS*NX*KNN, 32] bf16 = [edge(3), x(3), feat(24), 0, 0] ----------
__global__ __launch_bounds__(256) void gather_kernel(const float* __restrict__ p,
                                                     const float* __restrict__ pc,
                                                     const float* __restrict__ feat,
                                                     const int* __restrict__ idx,
                                                     __bf16* __restrict__ X) {
  size_t r = (size_t)blockIdx.x * 256 + threadIdx.x;
  if (r >= (size_t)BS * NX * KNN) return;
  int b = (int)(r / ((size_t)NX * KNN));
  size_t rem = r % ((size_t)NX * KNN);
  int n = (int)(rem / KNN);
  int j = idx[r];
  size_t pb = ((size_t)b * NX + n) * 3;
  size_t yb = ((size_t)b * NY + j) * 3;
  size_t fb = ((size_t)b * NY + j) * CD;
  float px = p[pb], py = p[pb + 1], pz = p[pb + 2];
  float yx = pc[yb], yy = pc[yb + 1], yz = pc[yb + 2];
  __bf16* o = X + r * 32;
  o[0] = (__bf16)(yx - px); o[1] = (__bf16)(yy - py); o[2] = (__bf16)(yz - pz);
  o[3] = (__bf16)px; o[4] = (__bf16)py; o[5] = (__bf16)pz;
#pragma unroll
  for (int c = 0; c < CD; ++c) o[6 + c] = (__bf16)feat[fb + c];
  o[30] = (__bf16)0.f; o[31] = (__bf16)0.f;
}

// ---------- GEMM (bf16 WMMA, f32 acc) + per-channel BN stats accumulation ----------
// X [nrows, KD] bf16 ; WT [ND][KD] bf16 (transposed) ; Y [nrows, ND] bf16 (raw pre-BN)
// XF: BN affine + LeakyReLU applied while staging A chunks (double-buffered).
// !XF: pure async DMA staging (KD must be 32).
template <int KD, int ND, bool XF, int NSTAT>
__global__ __launch_bounds__(256) void gemm_bn_kernel(const __bf16* __restrict__ X,
                                                      const __bf16* __restrict__ WT,
                                                      const float* __restrict__ sc,
                                                      const float* __restrict__ bi,
                                                      __bf16* __restrict__ Y,
                                                      float* __restrict__ gSum,
                                                      float* __restrict__ gSq) {
  constexpr int KT = KD / 32;
  __shared__ __attribute__((aligned(16))) __bf16 lA[2][128 * 32];
  __shared__ __attribute__((aligned(16))) __bf16 lW[ND * KD];
  __shared__ float sSum[ND], sSq[ND];
  __shared__ float sSc[KD], sBi[KD];
  int tid = threadIdx.x;
  size_t rowBase = (size_t)blockIdx.x * 128;

  // weights -> LDS via async DMA
  {
    unsigned lw0 = lds_off(lW);
    for (int e = tid; e < ND * KD / 8; e += 256)
      async_b128(lw0 + e * 16, (const char*)WT + e * 16);
  }
  if (tid < ND) { sSum[tid] = 0.f; sSq[tid] = 0.f; }
  if constexpr (XF) {
    for (int e = tid; e < KD; e += 256) { sSc[e] = sc[e]; sBi[e] = bi[e]; }
  }

  int lane = tid & 31, w = tid >> 5, m = lane & 15, g = lane >> 4;
  v8f acc[ND / 16];
#pragma unroll
  for (int nt = 0; nt < ND / 16; ++nt) acc[nt] = zero_v8f();

  if constexpr (!XF) {
    // single chunk (KD==32): A tile is contiguous -> async DMA it too
    unsigned la0 = lds_off(lA[0]);
    for (int e = tid; e < 128 * KD / 8; e += 256)
      async_b128(la0 + e * 16, (const char*)(X + rowBase * KD) + e * 16);
    wait_async0();
    __syncthreads();
    v16bf a = load_a_frag(&lA[0][(w * 16) * 32], m, g, 32);
#pragma unroll
    for (int nt = 0; nt < ND / 16; ++nt) {
      v16bf b = load_b_frag(lW, nt * 16 + m, g, 0, KD);
      acc[nt] = WMMA_BF16(a, b, acc[nt]);
    }
  } else {
    // double-buffered: prefetch next 128x32 chunk to regs while WMMA consumes LDS
    int r = tid >> 1;
    int k0 = (tid & 1) * 16;
    const __bf16* rowp = X + (rowBase + r) * KD + k0;
    union U { u32x4 q; __bf16 h[8]; };
    U c0, c1, n0, n1;
    c0.q = __builtin_nontemporal_load((const u32x4*)rowp);
    c1.q = __builtin_nontemporal_load((const u32x4*)rowp + 1);
    wait_async0();     // weights in LDS
    __syncthreads();   // weights + sSc/sBi visible
    int buf = 0;
#pragma unroll
    for (int kt = 0; kt < KT; ++kt) {
      U o0, o1;
#pragma unroll
      for (int i = 0; i < 8; ++i) {
        int ch = kt * 32 + k0 + i;
        float v = (float)c0.h[i];
        v = fmaf(v, sSc[ch], sBi[ch]);
        o0.h[i] = (__bf16)(v > 0.f ? v : 0.2f * v);
      }
#pragma unroll
      for (int i = 0; i < 8; ++i) {
        int ch = kt * 32 + k0 + 8 + i;
        float v = (float)c1.h[i];
        v = fmaf(v, sSc[ch], sBi[ch]);
        o1.h[i] = (__bf16)(v > 0.f ? v : 0.2f * v);
      }
      u32x4* dst = (u32x4*)&lA[buf][r * 32 + k0];
      dst[0] = o0.q;
      dst[1] = o1.q;
      if (kt + 1 < KT) {
        const u32x4* np = (const u32x4*)(rowp + (kt + 1) * 32);
        n0.q = __builtin_nontemporal_load(np);
        n1.q = __builtin_nontemporal_load(np + 1);
      }
      __syncthreads();
      v16bf a = load_a_frag(&lA[buf][(w * 16) * 32], m, g, 32);
#pragma unroll
      for (int nt = 0; nt < ND / 16; ++nt) {
        v16bf b = load_b_frag(lW, nt * 16 + m, g, kt * 32, KD);
        acc[nt] = WMMA_BF16(a, b, acc[nt]);
      }
      c0 = n0; c1 = n1;
      buf ^= 1;
    }
  }

  // epilogue: store raw pre-BN output + channel stats
#pragma unroll
  for (int nt = 0; nt < ND / 16; ++nt) {
    int nabs = nt * 16 + m;
    float s = 0.f, s2 = 0.f;
#pragma unroll
    for (int r = 0; r < 8; ++r) {
      float v = acc[nt][r];
      s += v; s2 += v * v;
      Y[(rowBase + w * 16 + r + 8 * g) * ND + nabs] = (__bf16)v;
    }
    atomicAdd(&sSum[nabs], s);
    atomicAdd(&sSq[nabs], s2);
  }
  __syncthreads();
  if (tid < NSTAT) {
    atomicAdd(&gSum[tid], sSum[tid]);
    atomicAdd(&gSq[tid], sSq[tid]);
  }
}

// ---------- BN finalize: per-channel scale/bias ----------
__global__ void bn_finalize_kernel(const float* __restrict__ sum, const float* __restrict__ sq,
                                   const float* __restrict__ gamma, const float* __restrict__ beta,
                                   float* __restrict__ sc, float* __restrict__ bi,
                                   int nch, float invc) {
  int c = threadIdx.x;
  if (c < nch) {
    float mu = sum[c] * invc;
    float var = sq[c] * invc - mu * mu;
    float a = gamma[c] * rsqrtf(var + 1e-5f);
    sc[c] = a;
    bi[c] = beta[c] - mu * a;
  } else {
    sc[c] = 0.f;
    bi[c] = 0.f;
  }
}

// ---------- BN3 + LeakyReLU + max over KNN neighbors -> cmax [BS*NX, 32] bf16 ----------
__global__ __launch_bounds__(256) void maxpool_kernel(const __bf16* __restrict__ Y3,
                                                      const float* __restrict__ sc,
                                                      const float* __restrict__ bi,
                                                      __bf16* __restrict__ cmax) {
  size_t t = (size_t)blockIdx.x * 256 + threadIdx.x;  // BS*NX*32
  int ch = (int)(t & 31);
  size_t pn = t >> 5;
  if (ch >= CD) { cmax[t] = (__bf16)0.f; return; }
  float a = sc[ch], b = bi[ch], mx = -3.4e38f;
#pragma unroll
  for (int k = 0; k < KNN; ++k) {
    float v = (float)Y3[(pn * KNN + k) * 32 + ch];
    v = fmaf(v, a, b);
    v = v > 0.f ? v : 0.2f * v;
    mx = fmaxf(mx, v);
  }
  cmax[t] = (__bf16)mx;
}

// ---------- fused ResNet-FC head: 128 points / block, net in registers, WMMA everywhere ----------
__global__ __launch_bounds__(256) void head_kernel(const float* __restrict__ p,
                                                   const __bf16* __restrict__ cmax,
                                                   const __bf16* __restrict__ wcT,  // [NB][128][32]
                                                   const float* __restrict__ bc,    // [NB][128]
                                                   const __bf16* __restrict__ w0T,  // [NB][128][128]
                                                   const float* __restrict__ b0,
                                                   const __bf16* __restrict__ w1T,
                                                   const float* __restrict__ b1,
                                                   const float* __restrict__ fpw,   // [3][128]
                                                   const float* __restrict__ fpb,
                                                   const float* __restrict__ fow,   // [128]
                                                   const float* __restrict__ fob,
                                                   float* __restrict__ occ) {
  __shared__ __attribute__((aligned(16))) __bf16 lW[128 * 128];
  __shared__ __attribute__((aligned(16))) __bf16 stage[8 * 16 * 32];
  __shared__ float red[8 * 16 * 16];
  int tid = threadIdx.x, lane = tid & 31, w = tid >> 5, m = lane & 15, g = lane >> 4;
  size_t rowBase = (size_t)blockIdx.x * 128;
  size_t r0 = rowBase + w * 16;
  __bf16* st = &stage[w * 512];
  unsigned lw0 = lds_off(lW);
  unsigned st0 = lds_off(st);

  float net[8][8];  // [ntile][r] ; element (row = r + 8*g, col = ntile*16 + m)
#pragma unroll
  for (int r = 0; r < 8; ++r) {
    size_t row = r0 + r + 8 * g;
    float px = p[row * 3 + 0], py = p[row * 3 + 1], pz = p[row * 3 + 2];
#pragma unroll
    for (int nt = 0; nt < 8; ++nt) {
      int n = nt * 16 + m;
      net[nt][r] = fpb[n] + px * fpw[0 * 128 + n] + py * fpw[1 * 128 + n] + pz * fpw[2 * 128 + n];
    }
  }

  for (int i = 0; i < NB; ++i) {
    // --- net += c @ Wc + bc --- (weights + per-wave cmax tile via async DMA)
    __syncthreads();
    for (int e = tid; e < 512; e += 256)
      async_b128(lw0 + e * 16, (const char*)(wcT + (size_t)i * 4096) + e * 16);
    for (int e = lane; e < 64; e += 32)
      async_b128(st0 + e * 16, (const char*)(cmax + r0 * 32) + e * 16);
    wait_async0();
    __syncthreads();
    {
      v16bf a = load_a_frag(st, m, g, 32);
#pragma unroll
      for (int nt = 0; nt < 8; ++nt) {
        v16bf b = load_b_frag(lW, nt * 16 + m, g, 0, 32);
        v8f acc = zero_v8f();
        acc = WMMA_BF16(a, b, acc);
        float bb = bc[i * 128 + nt * 16 + m];
#pragma unroll
        for (int r = 0; r < 8; ++r) net[nt][r] += acc[r] + bb;
      }
    }
    // --- hh = relu(net); a2 = hh @ W0 ---
    __syncthreads();
    for (int e = tid; e < 2048; e += 256)
      async_b128(lw0 + e * 16, (const char*)(w0T + (size_t)i * 16384) + e * 16);
    wait_async0();
    v8f a2[8];
#pragma unroll
    for (int nt = 0; nt < 8; ++nt) a2[nt] = zero_v8f();
#pragma unroll
    for (int kt = 0; kt < 4; ++kt) {
      __syncthreads();
#pragma unroll
      for (int half = 0; half < 2; ++half) {
        int nt = 2 * kt + half;
#pragma unroll
        for (int r = 0; r < 8; ++r) {
          float v = net[nt][r];
          v = v > 0.f ? v : 0.f;
          st[(r + 8 * g) * 32 + half * 16 + m] = (__bf16)v;
        }
      }
      __syncthreads();
      v16bf a = load_a_frag(st, m, g, 32);
#pragma unroll
      for (int nt = 0; nt < 8; ++nt) {
        v16bf b = load_b_frag(lW, nt * 16 + m, g, kt * 32, 128);
        a2[nt] = WMMA_BF16(a, b, a2[nt]);
      }
    }
    // rx = relu(a2 + b0)
    float rx[8][8];
#pragma unroll
    for (int nt = 0; nt < 8; ++nt) {
      float bb = b0[i * 128 + nt * 16 + m];
#pragma unroll
      for (int r = 0; r < 8; ++r) {
        float v = a2[nt][r] + bb;
        rx[nt][r] = v > 0.f ? v : 0.f;
      }
    }
    // --- a3 = rx @ W1 ; net += a3 + b1 ---
    __syncthreads();
    for (int e = tid; e < 2048; e += 256)
      async_b128(lw0 + e * 16, (const char*)(w1T + (size_t)i * 16384) + e * 16);
    wait_async0();
    v8f a3[8];
#pragma unroll
    for (int nt = 0; nt < 8; ++nt) a3[nt] = zero_v8f();
#pragma unroll
    for (int kt = 0; kt < 4; ++kt) {
      __syncthreads();
#pragma unroll
      for (int half = 0; half < 2; ++half) {
        int nt = 2 * kt + half;
#pragma unroll
        for (int r = 0; r < 8; ++r) st[(r + 8 * g) * 32 + half * 16 + m] = (__bf16)rx[nt][r];
      }
      __syncthreads();
      v16bf a = load_a_frag(st, m, g, 32);
#pragma unroll
      for (int nt = 0; nt < 8; ++nt) {
        v16bf b = load_b_frag(lW, nt * 16 + m, g, kt * 32, 128);
        a3[nt] = WMMA_BF16(a, b, a3[nt]);
      }
    }
#pragma unroll
    for (int nt = 0; nt < 8; ++nt) {
      float bb = b1[i * 128 + nt * 16 + m];
#pragma unroll
      for (int r = 0; r < 8; ++r) net[nt][r] += a3[nt][r] + bb;
    }
  }

  // --- occ = relu(net) @ fc_out_w + fc_out_b (LDS reduction across 16 lanes) ---
  float pr[8];
#pragma unroll
  for (int r = 0; r < 8; ++r) pr[r] = 0.f;
#pragma unroll
  for (int nt = 0; nt < 8; ++nt) {
    float wv = fow[nt * 16 + m];
#pragma unroll
    for (int r = 0; r < 8; ++r) {
      float v = net[nt][r];
      v = v > 0.f ? v : 0.f;
      pr[r] += v * wv;
    }
  }
  __syncthreads();
#pragma unroll
  for (int r = 0; r < 8; ++r) red[w * 256 + (r + 8 * g) * 16 + m] = pr[r];
  __syncthreads();
  if (lane < 16) {
    float s = fob[0];
#pragma unroll
    for (int j = 0; j < 16; ++j) s += red[w * 256 + lane * 16 + j];
    occ[r0 + lane] = s;
  }
}

// ------------------------------------------------------------------
extern "C" void kernel_launch(void* const* d_in, const int* in_sizes, int n_in,
                              void* d_out, int out_size, void* d_ws, size_t ws_size,
                              hipStream_t stream) {
  const float* p       = (const float*)d_in[0];
  const float* pc      = (const float*)d_in[1];
  const float* feat    = (const float*)d_in[2];
  const float* conv1_w = (const float*)d_in[3];
  const float* bn1_g   = (const float*)d_in[4];
  const float* bn1_b   = (const float*)d_in[5];
  const float* conv2_w = (const float*)d_in[6];
  const float* bn2_g   = (const float*)d_in[7];
  const float* bn2_b   = (const float*)d_in[8];
  const float* conv3_w = (const float*)d_in[9];
  const float* bn3_g   = (const float*)d_in[10];
  const float* bn3_b   = (const float*)d_in[11];
  const float* fc_p_w  = (const float*)d_in[12];
  const float* fc_p_b  = (const float*)d_in[13];
  const float* fc_c_w  = (const float*)d_in[14];
  const float* fc_c_b  = (const float*)d_in[15];
  const float* blk0_w  = (const float*)d_in[16];
  const float* blk0_b  = (const float*)d_in[17];
  const float* blk1_w  = (const float*)d_in[18];
  const float* blk1_b  = (const float*)d_in[19];
  const float* fc_out_w = (const float*)d_in[20];
  const float* fc_out_b = (const float*)d_in[21];
  float* occ = (float*)d_out;

  const size_t ROWS = (size_t)BS * NX * KNN;  // 655360
  char* ws = (char*)d_ws;
  size_t off = 0;
  auto alloc = [&](size_t bytes) -> void* {
    void* q = ws + off;
    off += (bytes + 255) & ~(size_t)255;
    return q;
  };

  int*    idx  = (int*)alloc((size_t)BS * NX * KNN * sizeof(int));
  __bf16* Xg   = (__bf16*)alloc(ROWS * 32 * 2);
  __bf16* Y1   = (__bf16*)alloc(ROWS * 128 * 2);
  __bf16* Y2   = (__bf16*)alloc(ROWS * 128 * 2);
  __bf16* Y3   = Y1;  // alias: Y1 dead after GEMM2
  __bf16* cmax = (__bf16*)alloc((size_t)BS * NX * 32 * 2);
  __bf16* w1T  = (__bf16*)alloc(128 * 32 * 2);
  __bf16* w2T  = (__bf16*)alloc(128 * 128 * 2);
  __bf16* w3T  = (__bf16*)alloc(32 * 128 * 2);
  __bf16* wcT  = (__bf16*)alloc((size_t)NB * 128 * 32 * 2);
  __bf16* w0T  = (__bf16*)alloc((size_t)NB * 128 * 128 * 2);
  __bf16* w1bT = (__bf16*)alloc((size_t)NB * 128 * 128 * 2);
  float* stats = (float*)alloc(6 * 128 * sizeof(float));
  float* sum1 = stats, *sq1 = stats + 128;
  float* sum2 = stats + 256, *sq2 = stats + 384;
  float* sum3 = stats + 512, *sq3 = stats + 640;
  float* sb = (float*)alloc((128 * 4 + 32 * 2) * sizeof(float));
  float* sc1 = sb, *bi1 = sb + 128, *sc2 = sb + 256, *bi2 = sb + 384;
  float* sc3 = sb + 512, *bi3 = sb + 544;

  (void)hipMemsetAsync(stats, 0, 6 * 128 * sizeof(float), stream);

  // weight conversion (transposed + padded bf16)
  cvt_t_kernel<<<(128 * 32 + 255) / 256, 256, 0, stream>>>(conv1_w, w1T, 30, 128, 32, 128);
  cvt_t_kernel<<<(128 * 128 + 255) / 256, 256, 0, stream>>>(conv2_w, w2T, 128, 128, 128, 128);
  cvt_t_kernel<<<(32 * 128 + 255) / 256, 256, 0, stream>>>(conv3_w, w3T, 128, CD, 128, 32);
  for (int i = 0; i < NB; ++i) {
    cvt_t_kernel<<<(128 * 32 + 255) / 256, 256, 0, stream>>>(fc_c_w + (size_t)i * CD * 128,
                                                             wcT + (size_t)i * 128 * 32, CD, 128, 32, 128);
    cvt_t_kernel<<<(128 * 128 + 255) / 256, 256, 0, stream>>>(blk0_w + (size_t)i * 128 * 128,
                                                              w0T + (size_t)i * 128 * 128, 128, 128, 128, 128);
    cvt_t_kernel<<<(128 * 128 + 255) / 256, 256, 0, stream>>>(blk1_w + (size_t)i * 128 * 128,
                                                              w1bT + (size_t)i * 128 * 128, 128, 128, 128, 128);
  }

  // KNN + gather
  knn_kernel<<<BS * (NX / 256), 256, 0, stream>>>(p, pc, idx);
  gather_kernel<<<(unsigned)((ROWS + 255) / 256), 256, 0, stream>>>(p, pc, feat, idx, Xg);

  const float invc = 1.f / (float)ROWS;
  unsigned gemmGrid = (unsigned)(ROWS / 128);  // 5120

  // layer 1: X(32) -> 128
  gemm_bn_kernel<32, 128, false, 128><<<gemmGrid, 256, 0, stream>>>(Xg, w1T, nullptr, nullptr, Y1, sum1, sq1);
  bn_finalize_kernel<<<1, 128, 0, stream>>>(sum1, sq1, bn1_g, bn1_b, sc1, bi1, 128, invc);
  // layer 2: bn+lrelu(Y1) (128) -> 128
  gemm_bn_kernel<128, 128, true, 128><<<gemmGrid, 256, 0, stream>>>(Y1, w2T, sc1, bi1, Y2, sum2, sq2);
  bn_finalize_kernel<<<1, 128, 0, stream>>>(sum2, sq2, bn2_g, bn2_b, sc2, bi2, 128, invc);
  // layer 3: bn+lrelu(Y2) (128) -> 24 (padded 32)
  gemm_bn_kernel<128, 32, true, 24><<<gemmGrid, 256, 0, stream>>>(Y2, w3T, sc2, bi2, Y3, sum3, sq3);
  bn_finalize_kernel<<<1, 32, 0, stream>>>(sum3, sq3, bn3_g, bn3_b, sc3, bi3, CD, invc);

  // bn3 + lrelu + max over neighbors
  maxpool_kernel<<<(unsigned)(((size_t)BS * NX * 32) / 256), 256, 0, stream>>>(Y3, sc3, bi3, cmax);

  // fused ResNet-FC head
  head_kernel<<<(BS * NX) / 128, 256, 0, stream>>>(p, cmax, wcT, fc_c_b, w0T, blk0_b, w1bT, blk1_b,
                                                   fc_p_w, fc_p_b, fc_out_w, fc_out_b, occ);
}
